// MCSTemporal_31061203484923
// MI455X (gfx1250) — compile-verified, matrix operand order checked by reading, not run on hardware
//
#include <hip/hip_runtime.h>
#include <hip/hip_bf16.h>
#include <math.h>

// ---------------- types ----------------
typedef __bf16 bf16_t;
typedef __attribute__((ext_vector_type(16))) __bf16 v16bf;
typedef __attribute__((ext_vector_type(8)))  float  v8f;

#define B_    128
#define S_    64
#define L_    5
#define FEAT_ 16
#define D_    128
#define DE_   128
#define MH_   256
#define M_    128
#define UH_   256
#define T_    64
#define EPER_ 512
#define N_    (2 * B_ * S_)     // 16384
#define E_    (2 * B_ * EPER_)  // 131072
#define LB_   (L_ * B_)         // 640

// ---------------- weight pack: f32 [K][Nc] -> bf16 WMMA B-fragments ----------------
// layout: frag index = ((kstep*NT + ntile)*32 + lane)*16 + i
// element (lane,i): k = kstep*32 + (i/8)*16 + (lane>>4)*8 + (i&7), n = ntile*16 + (lane&15)
__global__ void pack_b_kernel(const float* __restrict__ W, bf16_t* __restrict__ out,
                              int K, int Nc) {
    int tid = blockIdx.x * blockDim.x + threadIdx.x;
    int total = K * Nc;
    if (tid >= total) return;
    int i    = tid & 15;
    int lane = (tid >> 4) & 31;
    int rem  = tid >> 9;
    int NT   = Nc >> 4;
    int nt   = rem % NT;
    int ks   = rem / NT;
    int k = ks * 32 + ((i >> 3) << 4) + ((lane >> 4) << 3) + (i & 7);
    int n = nt * 16 + (lane & 15);
    out[tid] = (bf16_t)W[k * Nc + n];
}

// ---------------- encoders ----------------
__global__ void encode_nodes_kernel(const float* __restrict__ nf,
                                    const float* __restrict__ W, const float* __restrict__ b,
                                    float* __restrict__ h) {
    int tid = blockIdx.x * blockDim.x + threadIdx.x;
    if (tid >= N_ * D_) return;
    int n = tid >> 7, j = tid & 127;
    float acc = b[j];
    #pragma unroll
    for (int k = 0; k < FEAT_; ++k) acc += nf[n * FEAT_ + k] * W[k * D_ + j];
    h[tid] = acc;
}

__global__ void encode_edges_kernel(const float* __restrict__ ef,
                                    const float* __restrict__ W, const float* __restrict__ b,
                                    bf16_t* __restrict__ efb) {
    int tid = blockIdx.x * blockDim.x + threadIdx.x;
    if (tid >= E_ * DE_) return;
    int e = tid >> 7, j = tid & 127;
    float acc = b[j];
    #pragma unroll
    for (int k = 0; k < FEAT_; ++k) acc += ef[e * FEAT_ + k] * W[k * DE_ + j];
    efb[tid] = (bf16_t)acc;
}

// ---------------- WMMA fragment helpers ----------------
__device__ __forceinline__ v16bf load_a_frag(const bf16_t* __restrict__ Xrow,
                                             int kbase, int hi8) {
    v16bf a;
    #pragma unroll
    for (int i = 0; i < 16; ++i) {
        int k = kbase + ((i >> 3) << 4) + hi8 + (i & 7);
        a[i] = Xrow[k];
    }
    return a;
}

__device__ __forceinline__ v8f wmma_bf16(v16bf a, v16bf b, v8f c) {
    return __builtin_amdgcn_wmma_f32_16x16x32_bf16(false, a, false, b, (short)0, c,
                                                   false, false);
}

// ---------------- message MLP + scatter-add (one wave per 16-edge tile) ----------------
// x = [h[srcA], h[srcB], efb] (384)  -> relu(x@W1+b1)(256) -> (@W2+b2)(128) -> atomicAdd agg[dst]
__global__ __launch_bounds__(64) void message_kernel(
    const float* __restrict__ h, const bf16_t* __restrict__ efb,
    const int* __restrict__ srcA, const int* __restrict__ srcB, const int* __restrict__ dst,
    const bf16_t* __restrict__ W1p, const float* __restrict__ b1,
    const bf16_t* __restrict__ W2p, const float* __restrict__ b2,
    float* __restrict__ agg) {
    __shared__ bf16_t XS[2][16][384];
    __shared__ bf16_t YS[2][16][256];
    int wave = threadIdx.x >> 5;
    int lane = threadIdx.x & 31;
    int tile = blockIdx.x * 2 + wave;
    int e0 = tile * 16;
    if (e0 >= E_) return;

    // stage concat row in LDS as bf16 (wave-local; LDS ops are in-order per wave)
    for (int idx = lane; idx < 16 * 384; idx += 32) {
        int row = idx / 384, k = idx % 384;
        int e = e0 + row;
        float v;
        if (k < 128)      v = h[srcA[e] * D_ + k];
        else if (k < 256) v = h[srcB[e] * D_ + (k - 128)];
        else              v = (float)efb[(size_t)e * DE_ + (k - 256)];
        XS[wave][row][k] = (bf16_t)v;
    }

    const bf16_t* Xrow = &XS[wave][lane & 15][0];
    int hi8   = (lane >> 4) << 3;
    int rbase = hi8;  // C/D row base = 8*(lane>>4)

    // GEMM1: 16 N-tiles x 12 K-steps
    for (int nt = 0; nt < 16; ++nt) {
        v8f acc = {0.f, 0.f, 0.f, 0.f, 0.f, 0.f, 0.f, 0.f};
        #pragma unroll
        for (int ks = 0; ks < 12; ++ks) {
            v16bf a = load_a_frag(Xrow, ks * 32, hi8);
            v16bf b = *(const v16bf*)(W1p + (((size_t)(ks * 16 + nt) * 32 + lane) << 4));
            acc = wmma_bf16(a, b, acc);
        }
        int col = nt * 16 + (lane & 15);
        float bias = b1[col];
        #pragma unroll
        for (int v = 0; v < 8; ++v)
            YS[wave][rbase + v][col] = (bf16_t)fmaxf(acc[v] + bias, 0.f);
    }

    const bf16_t* Yrow = &YS[wave][lane & 15][0];
    int dsts[8];
    #pragma unroll
    for (int v = 0; v < 8; ++v) dsts[v] = dst[e0 + rbase + v];

    // GEMM2: 8 N-tiles x 8 K-steps, then scatter-add
    for (int nt = 0; nt < 8; ++nt) {
        v8f acc = {0.f, 0.f, 0.f, 0.f, 0.f, 0.f, 0.f, 0.f};
        #pragma unroll
        for (int ks = 0; ks < 8; ++ks) {
            v16bf a = load_a_frag(Yrow, ks * 32, hi8);
            v16bf b = *(const v16bf*)(W2p + (((size_t)(ks * 8 + nt) * 32 + lane) << 4));
            acc = wmma_bf16(a, b, acc);
        }
        int col = nt * 16 + (lane & 15);
        float bias = b2[col];
        #pragma unroll
        for (int v = 0; v < 8; ++v)
            atomicAdd(&agg[(size_t)dsts[v] * M_ + col], acc[v] + bias);
    }
}

// ---------------- node-update MLP (one wave per 16-node tile) ----------------
__global__ __launch_bounds__(64) void update_kernel(
    const float* __restrict__ agg, const float* __restrict__ hsrc,
    const bf16_t* __restrict__ W1p, const float* __restrict__ b1,
    const bf16_t* __restrict__ W2p, const float* __restrict__ b2,
    float* __restrict__ hdst, float* __restrict__ allh_l) {
    __shared__ bf16_t XS[2][16][256];
    __shared__ bf16_t YS[2][16][256];
    int wave = threadIdx.x >> 5;
    int lane = threadIdx.x & 31;
    int tile = blockIdx.x * 2 + wave;
    int n0 = tile * 16;
    if (n0 >= N_) return;

    for (int idx = lane; idx < 16 * 256; idx += 32) {
        int row = idx >> 8, k = idx & 255;
        int n = n0 + row;
        float v = (k < 128) ? agg[(size_t)n * M_ + k] : hsrc[(size_t)n * D_ + (k - 128)];
        XS[wave][row][k] = (bf16_t)v;
    }

    const bf16_t* Xrow = &XS[wave][lane & 15][0];
    int hi8   = (lane >> 4) << 3;
    int rbase = hi8;

    for (int nt = 0; nt < 16; ++nt) {
        v8f acc = {0.f, 0.f, 0.f, 0.f, 0.f, 0.f, 0.f, 0.f};
        #pragma unroll
        for (int ks = 0; ks < 8; ++ks) {
            v16bf a = load_a_frag(Xrow, ks * 32, hi8);
            v16bf b = *(const v16bf*)(W1p + (((size_t)(ks * 16 + nt) * 32 + lane) << 4));
            acc = wmma_bf16(a, b, acc);
        }
        int col = nt * 16 + (lane & 15);
        float bias = b1[col];
        #pragma unroll
        for (int v = 0; v < 8; ++v)
            YS[wave][rbase + v][col] = (bf16_t)fmaxf(acc[v] + bias, 0.f);
    }

    const bf16_t* Yrow = &YS[wave][lane & 15][0];
    for (int nt = 0; nt < 8; ++nt) {
        v8f acc = {0.f, 0.f, 0.f, 0.f, 0.f, 0.f, 0.f, 0.f};
        #pragma unroll
        for (int ks = 0; ks < 8; ++ks) {
            v16bf a = load_a_frag(Yrow, ks * 32, hi8);
            v16bf b = *(const v16bf*)(W2p + (((size_t)(ks * 8 + nt) * 32 + lane) << 4));
            acc = wmma_bf16(a, b, acc);
        }
        int col = nt * 16 + (lane & 15);
        float bias = b2[col];
        #pragma unroll
        for (int v = 0; v < 8; ++v) {
            float val = acc[v] + bias;
            size_t o = (size_t)(n0 + rbase + v) * D_ + col;
            hdst[o]   = val;
            allh_l[o] = val;
        }
    }
}

// ---------------- masked pad + set transform ----------------
// one block (128 thr) per (l,b,s,qc)
__global__ __launch_bounds__(128) void transform_kernel(
    const float* __restrict__ allh, const int* __restrict__ qsizes, const int* __restrict__ csizes,
    const float* __restrict__ Wt1, const float* __restrict__ bt1,
    const float* __restrict__ Wt2, const float* __restrict__ bt2,
    float* __restrict__ qpad, float* __restrict__ cpad,
    float* __restrict__ tq, float* __restrict__ tc) {
    int bid = blockIdx.x;
    int qc = bid & 1;
    int rid = bid >> 1;
    int s  = rid & 63;
    int lb = rid >> 6;          // l*B + b
    int b  = lb & 127;
    int l  = lb >> 7;
    int node = (2 * b + qc) * S_ + s;
    int size = qc ? csizes[b] : qsizes[b];
    float mask = (s < size) ? 1.f : 0.f;

    __shared__ float row[D_];
    __shared__ float hid[T_];
    int t = threadIdx.x;
    float v = allh[((size_t)l * N_ + node) * D_ + t] * mask;
    row[t] = v;
    (qc ? cpad : qpad)[((size_t)lb * S_ + s) * D_ + t] = v;
    __syncthreads();
    if (t < T_) {
        float a = bt1[t];
        #pragma unroll 4
        for (int k = 0; k < D_; ++k) a += row[k] * Wt1[k * T_ + t];
        hid[t] = fmaxf(a, 0.f);
    }
    __syncthreads();
    if (t < T_) {
        float a = bt2[t];
        #pragma unroll 4
        for (int k = 0; k < T_; ++k) a += hid[k] * Wt2[k * T_ + t];
        (qc ? tc : tq)[((size_t)lb * S_ + s) * T_ + t] = a * mask;
    }
}

// ---------------- sinkhorn + hinge score (one block per lb) ----------------
__global__ __launch_bounds__(256) void sinkhorn_score_kernel(
    const float* __restrict__ tq, const float* __restrict__ tc,
    const float* __restrict__ qpad, const float* __restrict__ cpad,
    float* __restrict__ scores) {
    extern __shared__ float sm[];
    float* P   = sm;          // 64*64
    float* buf = sm + 4096;   // up to 64*128
    __shared__ float red[256];
    int lb = blockIdx.x, t = threadIdx.x;

    for (int i = t; i < 4096; i += 256) buf[i] = tc[(size_t)lb * 4096 + i];
    __syncthreads();
    for (int i = t; i < 4096; i += 256) {
        int s = i >> 6, r = i & 63;
        const float* q = tq + (size_t)lb * 4096 + s * 64;
        const float* c = buf + r * 64;
        float d = 0.f;
        #pragma unroll 4
        for (int k = 0; k < 64; ++k) d += q[k] * c[k];
        P[i] = d * 10.0f;   // /TEMP
    }
    __syncthreads();
    for (int it = 0; it < 20; ++it) {
        if (t < 64) {  // normalize over r (axis 2)
            float mx = -INFINITY;
            for (int r = 0; r < 64; ++r) mx = fmaxf(mx, P[t * 64 + r]);
            float sum = 0.f;
            for (int r = 0; r < 64; ++r) sum += expf(P[t * 64 + r] - mx);
            float ls = mx + logf(sum);
            for (int r = 0; r < 64; ++r) P[t * 64 + r] -= ls;
        }
        __syncthreads();
        if (t < 64) {  // normalize over s (axis 1)
            float mx = -INFINITY;
            for (int s = 0; s < 64; ++s) mx = fmaxf(mx, P[s * 64 + t]);
            float sum = 0.f;
            for (int s = 0; s < 64; ++s) sum += expf(P[s * 64 + t] - mx);
            float ls = mx + logf(sum);
            for (int s = 0; s < 64; ++s) P[s * 64 + t] -= ls;
        }
        __syncthreads();
    }
    for (int i = t; i < 4096; i += 256) P[i] = expf(P[i]);
    __syncthreads();
    for (int i = t; i < 8192; i += 256) buf[i] = cpad[(size_t)lb * 8192 + i];
    __syncthreads();
    float local = 0.f;
    for (int i = t; i < 8192; i += 256) {
        int s = i >> 7, d = i & 127;
        float val = 0.f;
        #pragma unroll 4
        for (int r = 0; r < 64; ++r) val += P[s * 64 + r] * buf[r * 128 + d];
        float qv = qpad[(size_t)lb * 8192 + i];
        local += qv - fmaxf(qv - val, 0.f);
    }
    red[t] = local;
    __syncthreads();
    for (int off = 128; off > 0; off >>= 1) {
        if (t < off) red[t] += red[t + off];
        __syncthreads();
    }
    if (t == 0) scores[lb] = red[0];
}

__global__ void final_kernel(const float* __restrict__ scores,
                             const float* __restrict__ w, float* __restrict__ out) {
    int b = threadIdx.x;
    if (b >= B_) return;
    float a = 0.f;
    #pragma unroll
    for (int l = 0; l < L_; ++l) a += scores[l * B_ + b] * fmaxf(w[l], 0.f);
    out[b] = a;
}

// ---------------- host orchestration ----------------
extern "C" void kernel_launch(void* const* d_in, const int* in_sizes, int n_in,
                              void* d_out, int out_size, void* d_ws, size_t ws_size,
                              hipStream_t stream) {
    (void)in_sizes; (void)n_in; (void)out_size; (void)ws_size;
    const float* nf   = (const float*)d_in[0];
    const float* efr  = (const float*)d_in[1];
    const int*   from = (const int*)d_in[2];
    const int*   to   = (const int*)d_in[3];
    const int*   qsz  = (const int*)d_in[4];
    const int*   csz  = (const int*)d_in[5];
    const float* W_ne = (const float*)d_in[6];  const float* b_ne = (const float*)d_in[7];
    const float* W_ee = (const float*)d_in[8];  const float* b_ee = (const float*)d_in[9];
    const float* W_m1 = (const float*)d_in[10]; const float* b_m1 = (const float*)d_in[11];
    const float* W_m2 = (const float*)d_in[12]; const float* b_m2 = (const float*)d_in[13];
    const float* W_r1 = (const float*)d_in[14]; const float* b_r1 = (const float*)d_in[15];
    const float* W_r2 = (const float*)d_in[16]; const float* b_r2 = (const float*)d_in[17];
    const float* W_u1 = (const float*)d_in[18]; const float* b_u1 = (const float*)d_in[19];
    const float* W_u2 = (const float*)d_in[20]; const float* b_u2 = (const float*)d_in[21];
    const float* W_t1 = (const float*)d_in[22]; const float* b_t1 = (const float*)d_in[23];
    const float* W_t2 = (const float*)d_in[24]; const float* b_t2 = (const float*)d_in[25];
    const float* w_sc = (const float*)d_in[26];
    float* out = (float*)d_out;

    char* ws = (char*)d_ws;
    size_t off = 0;
    auto alloc = [&](size_t bytes) -> void* {
        void* p = ws + off;
        off = (off + bytes + 255) & ~(size_t)255;
        return p;
    };
    float*  h0    = (float*)alloc((size_t)N_ * D_ * 4);
    float*  h1    = (float*)alloc((size_t)N_ * D_ * 4);
    bf16_t* efb   = (bf16_t*)alloc((size_t)E_ * DE_ * 2);
    float*  allh  = (float*)alloc((size_t)L_ * N_ * D_ * 4);
    float*  agg   = (float*)alloc((size_t)N_ * M_ * 4);
    bf16_t* wm1p  = (bf16_t*)alloc((size_t)384 * 256 * 2);
    bf16_t* wm2p  = (bf16_t*)alloc((size_t)256 * 128 * 2);
    bf16_t* wr1p  = (bf16_t*)alloc((size_t)384 * 256 * 2);
    bf16_t* wr2p  = (bf16_t*)alloc((size_t)256 * 128 * 2);
    bf16_t* wu1p  = (bf16_t*)alloc((size_t)256 * 256 * 2);
    bf16_t* wu2p  = (bf16_t*)alloc((size_t)256 * 128 * 2);
    float*  qpad  = (float*)alloc((size_t)LB_ * S_ * D_ * 4);
    float*  cpad  = (float*)alloc((size_t)LB_ * S_ * D_ * 4);
    float*  tq    = (float*)alloc((size_t)LB_ * S_ * T_ * 4);
    float*  tc    = (float*)alloc((size_t)LB_ * S_ * T_ * 4);
    float*  score = (float*)alloc((size_t)LB_ * 4);

    auto packs = [&](const float* W, bf16_t* o, int K, int Nc) {
        int tot = K * Nc;
        pack_b_kernel<<<(tot + 255) / 256, 256, 0, stream>>>(W, o, K, Nc);
    };
    packs(W_m1, wm1p, 384, 256); packs(W_m2, wm2p, 256, 128);
    packs(W_r1, wr1p, 384, 256); packs(W_r2, wr2p, 256, 128);
    packs(W_u1, wu1p, 256, 256); packs(W_u2, wu2p, 256, 128);

    encode_nodes_kernel<<<(N_ * D_ + 255) / 256, 256, 0, stream>>>(nf, W_ne, b_ne, h0);
    encode_edges_kernel<<<(E_ * DE_ + 255) / 256, 256, 0, stream>>>(efr, W_ee, b_ee, efb);

    float* hsrc = h0;
    float* hdst = h1;
    int msg_blocks = (E_ / 16 + 1) / 2;   // 4096
    int upd_blocks = (N_ / 16 + 1) / 2;   // 512
    for (int l = 0; l < L_; ++l) {
        hipMemsetAsync(agg, 0, (size_t)N_ * M_ * 4, stream);
        // forward: concat(h[from], h[to], ef), scatter to to_idx
        message_kernel<<<msg_blocks, 64, 0, stream>>>(hsrc, efb, from, to, to,
                                                      wm1p, b_m1, wm2p, b_m2, agg);
        // reverse: concat(h[to], h[from], ef), scatter to from_idx
        message_kernel<<<msg_blocks, 64, 0, stream>>>(hsrc, efb, to, from, from,
                                                      wr1p, b_r1, wr2p, b_r2, agg);
        update_kernel<<<upd_blocks, 64, 0, stream>>>(agg, hsrc, wu1p, b_u1, wu2p, b_u2,
                                                     hdst, allh + (size_t)l * N_ * D_);
        float* tmp = hsrc; hsrc = hdst; hdst = tmp;
    }

    transform_kernel<<<LB_ * S_ * 2, 128, 0, stream>>>(allh, qsz, csz, W_t1, b_t1,
                                                       W_t2, b_t2, qpad, cpad, tq, tc);
    sinkhorn_score_kernel<<<LB_, 256, 48 * 1024, stream>>>(tq, tc, qpad, cpad, score);
    final_kernel<<<1, 128, 0, stream>>>(score, w_sc, out);
}
